// CoreAttention_37331855737189
// MI455X (gfx1250) — matrix operand order repaired and checked by reading, not compiled, not run on hardware
//
#include <hip/hip_runtime.h>

typedef __attribute__((ext_vector_type(16))) _Float16 v16h;
typedef __attribute__((ext_vector_type(8)))  _Float16 v8h;
typedef __attribute__((ext_vector_type(4)))  _Float16 v4h;
typedef __attribute__((ext_vector_type(8)))  float    v8f;

#define TILE_E 16

// ---------------------------------------------------------------------------
// Zero accumulators (output wV and workspace Z) -- must run every launch since
// the harness replays the graph without re-initializing buffers.
// ---------------------------------------------------------------------------
__global__ void ga_zero_kernel(float* __restrict__ out, int outN,
                               float* __restrict__ Zb, int zN) {
  int i = blockIdx.x * blockDim.x + threadIdx.x;
  if (i < outN) out[i] = 0.0f;
  if (i < zN)   Zb[i]  = 0.0f;
}

// ---------------------------------------------------------------------------
// One wave32 per 16-edge tile.
//  Phase 1: gather k[src] / q[dst] rows (f32) -> LDS (f16)
//  Phase 2: per head, V_WMMA_F32_16X16X32_F16; diag(D) = the 16 edge scores
//  Phase 3: scatter v[src]*score into out (atomics), score into Z
// ---------------------------------------------------------------------------
__global__ __launch_bounds__(32) void ga_edge_kernel(
    const float* __restrict__ q, const float* __restrict__ k,
    const float* __restrict__ v, const int* __restrict__ ei,
    float* __restrict__ out, float* __restrict__ Zb, int E) {
  __shared__ __align__(32) _Float16 kT[TILE_E * 128];
  __shared__ __align__(32) _Float16 qT[TILE_E * 128];
  __shared__ float sT[TILE_E * 8];

  const int lane = threadIdx.x;
  const int base = blockIdx.x * TILE_E;

  // lanes 0..15 fetch the tile's edge endpoints (clamped for tail tile)
  int sIdx = 0, dIdx = 0;
  if (lane < TILE_E) {
    int eg = base + lane;
    if (eg >= E) eg = E - 1;
    sIdx = ei[eg];       // src node
    dIdx = ei[E + eg];   // dst node
  }

  // ---- Phase 1: stage rows into LDS as f16 (coalesced 512B per row) ----
  for (int t = 0; t < TILE_E; ++t) {
    int se = __shfl(sIdx, t);
    int de = __shfl(dIdx, t);
    float4 kf = ((const float4*)(k + (size_t)se * 128))[lane];
    float4 qf = ((const float4*)(q + (size_t)de * 128))[lane];
    v4h kh, qh;
    kh[0] = (_Float16)kf.x; kh[1] = (_Float16)kf.y;
    kh[2] = (_Float16)kf.z; kh[3] = (_Float16)kf.w;
    qh[0] = (_Float16)qf.x; qh[1] = (_Float16)qf.y;
    qh[2] = (_Float16)qf.z; qh[3] = (_Float16)qf.w;
    *(v4h*)&kT[t * 128 + lane * 4] = kh;
    *(v4h*)&qT[t * 128 + lane * 4] = qh;
  }
  __syncthreads();

  // ---- Phase 2: scores via WMMA, one 16x16x32 f16 MMA per head ----
  // A (16x32 f16): lane<16 holds edge=lane, K=0..7 (head dims 0..7);
  //                lane>=16 holds edge=lane-16, K=8..15 (head dims 8..15);
  //                K=16..31 zero.
  // B (32x16 f16): lane<16 holds edge=lane, K=0..7 (head dims 0..7);
  //                lane>=16 holds edge=lane-16, K=8..15 (head dims 8..15); rest zero.
  // diag(D) = per-edge head score. Diag lives in lanes 0..7 (c[lane]) and
  // lanes 24..31 (c[lane-24]) per the 16x16 f32 C/D layout.
  const int  eA        = lane & 15;
  const int  hiSel     = (lane >> 4) & 1;
  const int  j         = (lane < 8) ? lane : ((lane - 24) & 7);
  const bool diagOwner = (lane < 8) || (lane >= 24);
  const int  eD        = (lane < 8) ? lane : (lane - 16);

#pragma unroll
  for (int h = 0; h < 8; ++h) {
    v16h a = {};
    v8h alo = *(const v8h*)&kT[eA * 128 + h * 16 + hiSel * 8];
#pragma unroll
    for (int i2 = 0; i2 < 8; ++i2) a[i2] = alo[i2];

    v16h b = {};
    v8h blo = *(const v8h*)&qT[eA * 128 + h * 16 + hiSel * 8];
#pragma unroll
    for (int i2 = 0; i2 < 8; ++i2) b[i2] = blo[i2];

    v8f c = {};
    c = __builtin_amdgcn_wmma_f32_16x16x32_f16(
        /*neg_a=*/false, a, /*neg_b=*/false, b,
        /*c_mod=*/(short)0, c, /*reuse_a=*/false, /*reuse_b=*/false);

    float dv = c[0];
#pragma unroll
    for (int r = 1; r < 8; ++r) dv = (j == r) ? c[r] : dv;

    if (diagOwner) {
      float x = dv * 0.25f;                 // / sqrt(d), d = 16
      x = fminf(fmaxf(x, -5.0f), 5.0f);
      sT[eD * 8 + h] = __expf(x);
    }
  }
  __syncthreads();

  // ---- Phase 3: scatter msg = v[src]*score and Z (L2 atomics) ----
  for (int e = 0; e < TILE_E; ++e) {
    if (base + e >= E) break;
    int se = __shfl(sIdx, e);
    int de = __shfl(dIdx, e);
    const float* vrow = v + (size_t)se * 128;
    float*       orow = out + (size_t)de * 128;
#pragma unroll
    for (int s2 = 0; s2 < 4; ++s2) {
      int idx = lane + 32 * s2;                // 0..127 over the row
      float sc = sT[e * 8 + (idx >> 4)];       // head = idx/16
      atomicAdd(&orow[idx], vrow[idx] * sc);
    }
    if (lane < 8) atomicAdd(&Zb[(size_t)de * 8 + lane], sT[e * 8 + lane]);
  }
}

// ---------------------------------------------------------------------------
// out = wV / (Z + 1e-6), in place.
// ---------------------------------------------------------------------------
__global__ void ga_norm_kernel(float* __restrict__ out,
                               const float* __restrict__ Zb, int total) {
  int i = blockIdx.x * blockDim.x + threadIdx.x;
  if (i >= total) return;
  int node = i >> 7;           // /128
  int head = (i >> 4) & 7;     // (i%128)/16
  out[i] = out[i] / (Zb[node * 8 + head] + 1e-6f);
}

extern "C" void kernel_launch(void* const* d_in, const int* in_sizes, int n_in,
                              void* d_out, int out_size, void* d_ws, size_t ws_size,
                              hipStream_t stream) {
  const float* q  = (const float*)d_in[0];
  const float* k  = (const float*)d_in[1];
  const float* v  = (const float*)d_in[2];
  const int*   ei = (const int*)d_in[3];   // [2, E] flattened

  const int nNodes = in_sizes[0] / 128;    // B*N
  const int E      = in_sizes[3] / 2;

  float* out = (float*)d_out;
  float* Zb  = (float*)d_ws;               // nNodes*8 floats
  const int zN = nNodes * 8;

  int mx = out_size > zN ? out_size : zN;
  ga_zero_kernel<<<(mx + 255) / 256, 256, 0, stream>>>(out, out_size, Zb, zN);

  int tiles = (E + TILE_E - 1) / TILE_E;
  ga_edge_kernel<<<tiles, 32, 0, stream>>>(q, k, v, ei, out, Zb, E);

  ga_norm_kernel<<<(out_size + 255) / 256, 256, 0, stream>>>(out, Zb, out_size);
}
